// MambaForecaster_89747636617268
// MI455X (gfx1250) — matrix operand order repaired and checked by reading, not compiled, not run on hardware
//
#include <hip/hip_runtime.h>
#include <hip/hip_bf16.h>
#include <math.h>

// ---------------- problem constants (match reference) ----------------
#define B_    32
#define L_    4096
#define DM    128      // d_model
#define DI    256      // d_inner
#define DS    16       // d_state
#define DTR   8        // dt_rank
#define NL    4        // layers
#define DCONV 4
#define NC    32       // scan chunks along L
#define CL    (L_/NC)  // 128 steps per chunk
#define M_TOT (B_*L_)  // 131072 rows for all GEMMs
#define DBLP  48       // x_proj output padded 40 -> 48 (3 WMMA n-tiles)

typedef __bf16 bf16_t;
typedef __attribute__((ext_vector_type(16))) __bf16 v16bf;
typedef __attribute__((ext_vector_type(8)))  __bf16 v8bf;
typedef __attribute__((ext_vector_type(8)))  float  v8f;
typedef __attribute__((ext_vector_type(4)))  unsigned int u32x4;
typedef __attribute__((ext_vector_type(8)))  int i32x8;
typedef __attribute__((ext_vector_type(4)))  int i32x4;

union V16U { v16bf v; v8bf h[2]; };

// ---- CDNA5 TDM availability (device pass only; host pass parses fallback) ----
#if defined(__has_builtin)
#  if __has_builtin(__builtin_amdgcn_tensor_load_to_lds)
#    define HAVE_TDM 1
#  endif
#  if __has_builtin(__builtin_amdgcn_s_wait_tensorcnt)
#    define HAVE_TCNT 1
#  endif
#endif
#ifndef HAVE_TDM
#define HAVE_TDM 0
#endif
#ifndef HAVE_TCNT
#define HAVE_TCNT 0
#endif
#if __has_include(<hip/amd_detail/amd_gfx1250_TDM.h>)
#define TDM_6ARG 1   // therock headers -> clang-23 6-arg builtin
#else
#define TDM_6ARG 0   // ROCm 7.2 -> 5-arg builtin
#endif

__device__ __forceinline__ float sigmoidf_(float x){ return 1.f/(1.f+__expf(-x)); }
__device__ __forceinline__ float softplusf_(float x){ return x > 2e1f ? x : log1pf(__expf(x)); }

// Issue one TDM 2D tile load: rows x K bf16 elements, global row stride = K elems,
// LDS destination padded +16B per row (pad_interval = one row, pad_amount = 4 DWORDs).
// D# bit layout per CDNA5 ISA ch.8 (group0: count/lds_addr/global_addr/type,
// group1: data_size/pad/tensor dims/tile dims/stride).
template<int K>
__device__ __forceinline__ void tdm_load_2d(unsigned lds_off, const bf16_t* gptr, unsigned rows)
{
#if HAVE_TDM
    constexpr unsigned pad_iv = (K == 128) ? 5u : 6u;   // 64 / 128 DWORDs between pads (one row)
    unsigned long long ga = (unsigned long long)(uintptr_t)gptr;
    u32x4 g0;
    g0[0] = 0x1u;                                        // count=1, user descriptor
    g0[1] = lds_off;                                     // lds_addr (bytes)
    g0[2] = (unsigned)(ga & 0xffffffffull);              // global_addr[31:0]
    g0[3] = (unsigned)((ga >> 32) & 0x01ffffffull) | (2u << 30);  // addr[56:32] | type=2
    i32x8 g1;
    g1[0] = (int)((1u << 16) | (1u << 20) | (pad_iv << 22) | (3u << 25));
        // data_size=1(2B) | pad_enable | pad_interval | pad_amount=4 DWORDs (16B)
    g1[1] = (int)(((unsigned)K & 0xffffu) << 16);        // tensor_dim0[15:0]
    g1[2] = (int)((((unsigned)K >> 16) & 0xffffu) | ((rows & 0xffffu) << 16)); // dim0[31:16]|dim1[15:0]
    g1[3] = (int)(((rows >> 16) & 0xffffu) | (((unsigned)K & 0xffffu) << 16)); // dim1[31:16]|tile_dim0
    g1[4] = (int)(rows & 0xffffu);                       // tile_dim1 (tile_dim2=0)
    g1[5] = (int)K;                                      // tensor_dim0_stride[31:0]
    g1[6] = 0; g1[7] = 0;
    i32x4 z4 = {0,0,0,0};
#if TDM_6ARG
    i32x8 z8 = {0,0,0,0,0,0,0,0};
    __builtin_amdgcn_tensor_load_to_lds(g0, g1, z4, z4, z8, 0);
#else
    __builtin_amdgcn_tensor_load_to_lds(g0, g1, z4, z4, 0);
#endif
#else
    (void)lds_off; (void)gptr; (void)rows;
#endif
}

// ---------------- weight precasts ----------------
__global__ void cast_kernel(const float* __restrict__ src, bf16_t* __restrict__ dst, int n){
    int i = blockIdx.x*blockDim.x + threadIdx.x;
    if (i < n) dst[i] = (bf16_t)src[i];
}

// x_proj_w [NL][40][256] -> bf16 [NL][48][256], rows 40..47 zero
__global__ void pad_xproj_kernel(const float* __restrict__ src, bf16_t* __restrict__ dst){
    int idx = blockIdx.x*blockDim.x + threadIdx.x;   // NL*48*DI
    int k = idx % DI;
    int n = (idx / DI) % DBLP;
    int l = idx / (DI*DBLP);
    float v = (n < (DTR + 2*DS)) ? src[((size_t)l*(DTR+2*DS) + n)*DI + k] : 0.f;
    dst[idx] = (bf16_t)v;
}

// ---------------- h = x * input_proj_w + b ----------------
__global__ void init_h_kernel(const float* __restrict__ x, const float* __restrict__ ipw,
                              const float* __restrict__ ipb, float* __restrict__ h,
                              bf16_t* __restrict__ hbf){
    size_t idx = (size_t)blockIdx.x*blockDim.x + threadIdx.x;  // B*L*DM
    int d = (int)(idx % DM);
    size_t bl = idx / DM;
    float v = x[bl]*ipw[d] + ipb[d];
    h[idx] = v; hbf[idx] = (bf16_t)v;
}

// ---------------- bf16 WMMA GEMM with TDM->LDS panel staging ----------------
// Block = 4 waves = 64 rows x (16*NT) cols. A-panel (64xK) and W-panel (16*NT x K)
// are DMA'd to LDS with +16B/row padding, then WMMAs are fed from ds loads.
// C[M,N] (+= residual if RESID) = A[M,K] * W[N,K]^T
template<int NT, int K, bool RESID>
__global__ __launch_bounds__(128)
void gemm_wmma_kernel(const bf16_t* __restrict__ A, const bf16_t* __restrict__ W,
                      float* __restrict__ C, bf16_t* __restrict__ Cbf, int M, int N)
{
    constexpr int KP = K + 8;                 // padded LDS row stride (elements)
    constexpr int WROWS = 16*NT;
    __shared__ __align__(32) bf16_t ldsA[64*KP];
    __shared__ __align__(32) bf16_t ldsW[WROWS*KP];

    const int nstrips = N / (16*NT);
    const int bm = blockIdx.x / nstrips;      // 64-row block tile
    const int ns = blockIdx.x % nstrips;      // n-strip
    const int wv   = threadIdx.x >> 5;
    const int lane = threadIdx.x & 31;
    const int row  = lane & 15;
    const int grp  = lane >> 4;

    const bf16_t* Ag = A + (size_t)bm*64*K;
    const bf16_t* Wg = W + (size_t)ns*WROWS*K;

#if HAVE_TDM
    if (wv == 0) {
        tdm_load_2d<K>((unsigned)(uintptr_t)&ldsA[0], Ag, 64u);
        tdm_load_2d<K>((unsigned)(uintptr_t)&ldsW[0], Wg, (unsigned)WROWS);
#if HAVE_TCNT
        __builtin_amdgcn_s_wait_tensorcnt(0);
#endif
    }
    __syncthreads();
#else
    for (int i = threadIdx.x; i < 64*(K/8); i += 128) {
        int r = i/(K/8), cc = i%(K/8);
        *(v8bf*)(ldsA + r*KP + cc*8) = *(const v8bf*)(Ag + (size_t)r*K + cc*8);
    }
    for (int i = threadIdx.x; i < WROWS*(K/8); i += 128) {
        int r = i/(K/8), cc = i%(K/8);
        *(v8bf*)(ldsW + r*KP + cc*8) = *(const v8bf*)(Wg + (size_t)r*K + cc*8);
    }
    __syncthreads();
#endif

    v8f acc[NT];
    #pragma unroll
    for (int t = 0; t < NT; t++) {
        #pragma unroll
        for (int r = 0; r < 8; r++) {
            if (RESID) {
                int m = bm*64 + wv*16 + r + grp*8;
                int n = (ns*NT + t)*16 + row;
                acc[t][r] = C[(size_t)m*N + n];
            } else {
                acc[t][r] = 0.f;
            }
        }
    }

    const bf16_t* Arow = ldsA + (wv*16 + row)*KP;
    #pragma unroll
    for (int kt = 0; kt < K/32; kt++) {
        // A 16x32 bf16 tile regs: lanes 0-15 rows hold K {0..7,16..23}; lanes 16-31 K {8..15,24..31}
        V16U au;
        au.h[0] = *(const v8bf*)(Arow + kt*32 + grp*8);
        au.h[1] = *(const v8bf*)(Arow + kt*32 + 16 + grp*8);
        v16bf a = au.v;
        #pragma unroll
        for (int t = 0; t < NT; t++) {
            // B 32x16 tile: column (lane&15); lanes 0-15 hold K {0..7,16..23}; lanes 16-31 K {8..15,24..31}
            const bf16_t* wrow = ldsW + (t*16 + row)*KP + kt*32 + grp*8;
            V16U bu;
            bu.h[0] = *(const v8bf*)wrow;
            bu.h[1] = *(const v8bf*)(wrow + 16);
            acc[t] = __builtin_amdgcn_wmma_f32_16x16x32_bf16(
                false, a, false, bu.v, (short)0, acc[t], false, false);
        }
    }

    #pragma unroll
    for (int t = 0; t < NT; t++) {
        #pragma unroll
        for (int r = 0; r < 8; r++) {
            int m = bm*64 + wv*16 + r + grp*8;
            int n = (ns*NT + t)*16 + row;
            float val = acc[t][r];
            C[(size_t)m*N + n] = val;
            if (RESID) Cbf[(size_t)m*N + n] = (bf16_t)val;
        }
    }
}

// ---------------- depthwise causal conv (k=4) + bias + silu ----------------
__global__ void conv_silu_kernel(const float* __restrict__ xz, const float* __restrict__ cw,
                                 const float* __restrict__ cb, float* __restrict__ xc,
                                 bf16_t* __restrict__ xcbf){
    size_t idx = (size_t)blockIdx.x*blockDim.x + threadIdx.x;  // B*L*DI
    int c = (int)(idx % DI);
    size_t bt = idx / DI;
    int t = (int)(bt % L_);
    size_t brow = bt - t;                                       // b*L
    float acc = cb[c];
    #pragma unroll
    for (int j = 0; j < DCONV; j++) {
        int tt = t + j - (DCONV - 1);
        if (tt >= 0) acc += cw[c*DCONV + j] * xz[(brow + tt)*(2*DI) + c];
    }
    float v = acc * sigmoidf_(acc);
    xc[idx] = v; xcbf[idx] = (bf16_t)v;
}

// ---------------- dt = softplus(dbl[:,:8] @ dtw^T + b) ----------------
__global__ void dt_kernel(const float* __restrict__ dbl, const float* __restrict__ dtw,
                          const float* __restrict__ dtbias, float* __restrict__ dtb){
    size_t idx = (size_t)blockIdx.x*blockDim.x + threadIdx.x;  // B*L*DI
    int c = (int)(idx % DI);
    size_t bt = idx / DI;
    float acc = dtbias[c];
    #pragma unroll
    for (int r = 0; r < DTR; r++) acc += dbl[bt*DBLP + r] * dtw[c*DTR + r];
    dtb[idx] = softplusf_(acc);
}

// ---------------- chunked selective scan ----------------
// pass1: per (b,chunk,channel) compute P = prod exp(dt*A), F = chunk-final h with h0=0
__global__ __launch_bounds__(256)
void scan_pass1_kernel(const float* __restrict__ dbl, const float* __restrict__ dtb,
                       const float* __restrict__ Alog, const float* __restrict__ xc,
                       float* __restrict__ P, float* __restrict__ F){
    int b = blockIdx.x / NC, k = blockIdx.x % NC;
    int c = threadIdx.x;                 // 256 channels
    int lane = c & 31;
    float a[DS], h[DS], pr[DS];
    #pragma unroll
    for (int s = 0; s < DS; s++) { a[s] = -__expf(Alog[c*DS + s]); h[s] = 0.f; pr[s] = 1.f; }
    for (int t = k*CL; t < (k+1)*CL; t++) {
        size_t base = (size_t)b*L_ + t;
        float bc  = dbl[base*DBLP + DTR + lane];   // lanes 0-15: B_t, 16-31: C_t (unused here)
        float dtv = dtb[base*DI + c];
        float xv  = xc [base*DI + c];
        float dx = dtv*xv;
        #pragma unroll
        for (int s = 0; s < DS; s++) {
            float Bs = __shfl(bc, s, 32);
            float e = __expf(dtv*a[s]);
            h[s] = e*h[s] + dx*Bs;
            pr[s] *= e;
        }
    }
    size_t o = ((size_t)(b*NC + k)*DI + c)*DS;
    #pragma unroll
    for (int s = 0; s < DS; s++) { P[o + s] = pr[s]; F[o + s] = h[s]; }
}

// pass2: prefix over chunks -> H0 (initial state per chunk)
__global__ void scan_pass2_kernel(const float* __restrict__ P, const float* __restrict__ F,
                                  float* __restrict__ H0){
    int tid = blockIdx.x*blockDim.x + threadIdx.x;   // B*DI*DS
    int b = tid / (DI*DS);
    int rem = tid % (DI*DS);
    int c = rem / DS, s = rem % DS;
    float st = 0.f;
    for (int k = 0; k < NC; k++) {
        size_t o = ((size_t)(b*NC + k)*DI + c)*DS + s;
        H0[o] = st;
        st = P[o]*st + F[o];
    }
}

// pass3: re-run with correct h0, emit gated output y_g = (y + D*x) * silu(z) as bf16
__global__ __launch_bounds__(256)
void scan_pass3_kernel(const float* __restrict__ dbl, const float* __restrict__ dtb,
                       const float* __restrict__ Alog, const float* __restrict__ H0,
                       const float* __restrict__ xc, const float* __restrict__ xz,
                       const float* __restrict__ Dp, bf16_t* __restrict__ yg){
    int b = blockIdx.x / NC, k = blockIdx.x % NC;
    int c = threadIdx.x;
    int lane = c & 31;
    float a[DS], h[DS];
    size_t o = ((size_t)(b*NC + k)*DI + c)*DS;
    #pragma unroll
    for (int s = 0; s < DS; s++) { a[s] = -__expf(Alog[c*DS + s]); h[s] = H0[o + s]; }
    float Dc = Dp[c];
    for (int t = k*CL; t < (k+1)*CL; t++) {
        size_t base = (size_t)b*L_ + t;
        float bc  = dbl[base*DBLP + DTR + lane];   // lanes 0-15: B_t, lanes 16-31: C_t
        float dtv = dtb[base*DI + c];
        float xv  = xc [base*DI + c];
        float zv  = xz [base*(2*DI) + DI + c];
        float dx = dtv*xv;
        float y = 0.f;
        #pragma unroll
        for (int s = 0; s < DS; s++) {
            float Bs = __shfl(bc, s, 32);
            float Cs = __shfl(bc, 16 + s, 32);
            h[s] = __expf(dtv*a[s])*h[s] + dx*Bs;
            y += h[s]*Cs;
        }
        float out = (y + Dc*xv) * (zv * sigmoidf_(zv));
        yg[base*DI + c] = (bf16_t)out;
    }
}

// ---------------- RMSNorm(last token) + output projection ----------------
__global__ void final_kernel(const float* __restrict__ h, const float* __restrict__ nw,
                             const float* __restrict__ opw, const float* __restrict__ opb,
                             float* __restrict__ out){
    int b = blockIdx.x, d = threadIdx.x;     // 128 threads
    __shared__ float red[DM];
    float v = h[((size_t)b*L_ + (L_-1))*DM + d];
    red[d] = v*v; __syncthreads();
    for (int off = DM/2; off > 0; off >>= 1) { if (d < off) red[d] += red[d+off]; __syncthreads(); }
    float rms = sqrtf(red[0]/(float)DM + 1e-6f);
    __syncthreads();
    red[d] = (nw[d]*v/rms)*opw[d]; __syncthreads();
    for (int off = DM/2; off > 0; off >>= 1) { if (d < off) red[d] += red[d+off]; __syncthreads(); }
    if (d == 0) out[b] = red[0] + opb[0];
}

// ---------------- host launcher ----------------
extern "C" void kernel_launch(void* const* d_in, const int* in_sizes, int n_in,
                              void* d_out, int out_size, void* d_ws, size_t ws_size,
                              hipStream_t stream)
{
    const float* x    = (const float*)d_in[0];
    const float* ipw  = (const float*)d_in[1];
    const float* ipb  = (const float*)d_in[2];
    const float* inw  = (const float*)d_in[3];
    const float* cw   = (const float*)d_in[4];
    const float* cb   = (const float*)d_in[5];
    const float* xpw  = (const float*)d_in[6];
    const float* dtw  = (const float*)d_in[7];
    const float* dtbi = (const float*)d_in[8];
    const float* alog = (const float*)d_in[9];
    const float* Dp   = (const float*)d_in[10];
    const float* ow   = (const float*)d_in[11];
    const float* nw   = (const float*)d_in[12];
    const float* opw  = (const float*)d_in[13];
    const float* opb  = (const float*)d_in[14];
    float* out = (float*)d_out;
    (void)in_sizes; (void)n_in; (void)out_size; (void)ws_size;

    char* p = (char*)d_ws;
    auto take = [&](size_t n){ char* r = p; p += (n + 255) & ~(size_t)255; return r; };
    float*  h    = (float*) take((size_t)M_TOT*DM*4);
    bf16_t* hbf  = (bf16_t*)take((size_t)M_TOT*DM*2);
    float*  xz   = (float*) take((size_t)M_TOT*2*DI*4);
    float*  xc   = (float*) take((size_t)M_TOT*DI*4);
    bf16_t* xcbf = (bf16_t*)take((size_t)M_TOT*DI*2);
    float*  dbl  = (float*) take((size_t)M_TOT*DBLP*4);
    float*  dtb  = (float*) take((size_t)M_TOT*DI*4);
    bf16_t* yg   = (bf16_t*)take((size_t)M_TOT*DI*2);
    float*  P    = (float*) take((size_t)B_*NC*DI*DS*4);
    float*  F    = (float*) take((size_t)B_*NC*DI*DS*4);
    float*  H0   = (float*) take((size_t)B_*NC*DI*DS*4);
    bf16_t* winb = (bf16_t*)take((size_t)NL*2*DI*DM*2);
    bf16_t* wxpb = (bf16_t*)take((size_t)NL*DBLP*DI*2);
    bf16_t* wob  = (bf16_t*)take((size_t)NL*DM*DI*2);

    // weight precasts (fp32 -> bf16, x_proj zero-padded 40->48 rows)
    cast_kernel<<<(NL*2*DI*DM)/256, 256, 0, stream>>>(inw, winb, NL*2*DI*DM);
    pad_xproj_kernel<<<(NL*DBLP*DI)/256, 256, 0, stream>>>(xpw, wxpb);
    cast_kernel<<<(NL*DM*DI)/256, 256, 0, stream>>>(ow, wob, NL*DM*DI);

    init_h_kernel<<<(M_TOT*DM)/256, 256, 0, stream>>>(x, ipw, ipb, h, hbf);

    for (int i = 0; i < NL; i++) {
        // xz = h @ in_proj_w^T      [M,512] = [M,128]x[512,128]^T ; NT=4 -> 8 n-strips
        gemm_wmma_kernel<4, DM, false>
            <<<(M_TOT/64)*(2*DI/64), 128, 0, stream>>>(hbf, winb + (size_t)i*2*DI*DM, xz, nullptr, M_TOT, 2*DI);
        // depthwise causal conv + silu
        conv_silu_kernel<<<(M_TOT*DI)/256, 256, 0, stream>>>(xz, cw + i*DI*DCONV, cb + i*DI, xc, xcbf);
        // dbl = xc @ x_proj_w^T     [M,48] = [M,256]x[48,256]^T (padded) ; NT=3 -> 1 strip
        gemm_wmma_kernel<3, DI, false>
            <<<(M_TOT/64)*(DBLP/48), 128, 0, stream>>>(xcbf, wxpb + (size_t)i*DBLP*DI, dbl, nullptr, M_TOT, DBLP);
        // dt = softplus(dbl[:,:8] @ dt_proj_w^T + b)
        dt_kernel<<<(M_TOT*DI)/256, 256, 0, stream>>>(dbl, dtw + i*DI*DTR, dtbi + i*DI, dtb);
        // chunked selective scan (3 passes)
        scan_pass1_kernel<<<B_*NC, 256, 0, stream>>>(dbl, dtb, alog + i*DI*DS, xc, P, F);
        scan_pass2_kernel<<<(B_*DI*DS)/256, 256, 0, stream>>>(P, F, H0);
        scan_pass3_kernel<<<B_*NC, 256, 0, stream>>>(dbl, dtb, alog + i*DI*DS, H0, xc, xz, Dp + i*DI, yg);
        // h += yg @ out_proj_w^T    [M,128] += [M,256]x[128,256]^T ; NT=2 -> 4 n-strips
        gemm_wmma_kernel<2, DI, true>
            <<<(M_TOT/64)*(DM/32), 128, 0, stream>>>(yg, wob + (size_t)i*DM*DI, h, hbf, M_TOT, DM);
    }

    final_kernel<<<B_, DM, 0, stream>>>(h, nw, opw, opb, out);
}
